// pt_bn_CrossLayerLight_25220047962586
// MI455X (gfx1250) — matrix operand order repaired and checked by reading, not compiled, not run on hardware
//
#include <hip/hip_runtime.h>

#define NPT   4096
#define BATCH 2
#define NPALL (BATCH * NPT)   // 8192 points total
#define EPSV  1e-5f
#define LEAKV 0.1f

typedef __attribute__((ext_vector_type(8)))  float  v8f;
typedef __attribute__((ext_vector_type(16))) __bf16 v16bf;
typedef __attribute__((ext_vector_type(8)))  __bf16 v8bf;

// ---------------------------------------------------------------------------
// WMMA helpers: D(16x16,f32) += A(16x32,bf16) x B(32x16,bf16)
// Fragment layouts: contiguous K per lane half, k = 16*(lane/16) + s, in both A and B (wave32).
// ---------------------------------------------------------------------------
__device__ __forceinline__ v8f wmma_bf16(v16bf a, v16bf b, v8f c) {
  return __builtin_amdgcn_wmma_f32_16x16x32_bf16(false, a, false, b, (short)0, c,
                                                 false, false);
}

// A fragment from LDS weight matrix W (row-major, 64 columns = K).
// lane<16: row = mtile*16+lane, K = kbase+0..15
// lane>=16: row = mtile*16+lane-16, K = kbase+16..31
__device__ __forceinline__ v16bf load_afrag(const __bf16* W, int mtile, int kstep,
                                            int lane) {
  int row = mtile * 16 + (lane & 15);
  int kb  = kstep * 32 + ((lane >> 4) << 4);
  const __bf16* p = W + row * 64 + kb;
  return *(const v16bf*)(p);
}

// B fragment from LDS activation tile X, point-major: X[col][64 channels].
// lane<16: col = lane, K = kbase+0..15 ; lane>=16: col = lane-16, K = kbase+16..31
__device__ __forceinline__ v16bf load_bfrag(const __bf16* X, int kstep, int lane) {
  int col = lane & 15;
  int kb  = kstep * 32 + ((lane >> 4) << 4);
  return *(const v16bf*)(X + col * 64 + kb);
}

// Out(64x16) = W(64x64) @ X(64x16): 4 M-tiles x 2 K-steps = 8 WMMAs.
__device__ __forceinline__ void gemm_64x64x16(const __bf16* W, const __bf16* X,
                                              int lane, v8f acc[4]) {
  v16bf b0 = load_bfrag(X, 0, lane);
  v16bf b1 = load_bfrag(X, 1, lane);
#pragma unroll
  for (int t = 0; t < 4; ++t) {
    v16bf a0 = load_afrag(W, t, 0, lane);
    v16bf a1 = load_afrag(W, t, 1, lane);
    acc[t] = wmma_bf16(a0, b0, acc[t]);
    acc[t] = wmma_bf16(a1, b1, acc[t]);
  }
}

__device__ __forceinline__ void zero_acc4(v8f acc[4]) {
#pragma unroll
  for (int t = 0; t < 4; ++t)
#pragma unroll
    for (int r = 0; r < 8; ++r) acc[t][r] = 0.0f;
}

// ---------------------------------------------------------------------------
// KNN: per-thread insertion-sorted top-16 (ascending dist) over LDS ref tiles.
// ---------------------------------------------------------------------------
__global__ void __launch_bounds__(256)
knn16_kernel(const float* __restrict__ pcq, const float* __restrict__ pcr,
             int* __restrict__ idx) {
  __shared__ float rx[512], ry[512], rz[512];
  int tid = threadIdx.x;
  int gp  = blockIdx.x * 256 + tid;            // global query point
  int b   = gp >> 12;
  int n   = gp & (NPT - 1);
  const float* q = pcq + b * 3 * NPT;          // pc layout: (B,3,N)
  const float* r = pcr + b * 3 * NPT;
  float qx = q[n], qy = q[NPT + n], qz = q[2 * NPT + n];
  float bd[16];
  int   bi[16];
#pragma unroll
  for (int s = 0; s < 16; ++s) { bd[s] = 3.0e38f; bi[s] = 0; }

  for (int t0 = 0; t0 < NPT; t0 += 512) {
    __syncthreads();
    for (int e = tid; e < 512; e += 256) {
      rx[e] = r[t0 + e];
      ry[e] = r[NPT + t0 + e];
      rz[e] = r[2 * NPT + t0 + e];
    }
    __syncthreads();
    for (int j = 0; j < 512; ++j) {
      float dx = qx - rx[j], dy = qy - ry[j], dz = qz - rz[j];
      float d  = dx * dx + dy * dy + dz * dz;
      if (d < bd[15]) {
        int ri = t0 + j;
#pragma unroll
        for (int s = 0; s < 16; ++s) {
          if (d < bd[s]) {
            float td = bd[s]; bd[s] = d; d = td;
            int   ti = bi[s]; bi[s] = ri; ri = ti;
          }
        }
      }
    }
  }
#pragma unroll
  for (int s = 0; s < 16; ++s) idx[gp * 16 + s] = bi[s];
}

// ---------------------------------------------------------------------------
// tlinear: out = relu(bn(W(64x64) @ feat + bias)); feat (B,64,N) channel-major,
// out point-major (B*N,64). One wave -> 16 points.
// ---------------------------------------------------------------------------
__global__ void __launch_bounds__(256)
tlinear_kernel(const float* __restrict__ feat, const float* __restrict__ W,
               const float* __restrict__ bias, const float* __restrict__ g,
               const float* __restrict__ bb, float* __restrict__ outPM) {
  __shared__ alignas(32) __bf16 Wl[64 * 64];
  __shared__ alignas(32) __bf16 Xl[128 * 64];
  __shared__ float sS[64], shS[64];
  int tid = threadIdx.x;
  const float rs = rsqrtf(1.0f + EPSV);
  for (int e = tid; e < 4096; e += 256) Wl[e] = (__bf16)W[e];
  if (tid < 64) {
    float s  = g[tid] * rs;
    sS[tid]  = s;
    shS[tid] = bias[tid] * s + bb[tid];
  }
  int pbase = blockIdx.x * 128;
  int b     = pbase >> 12;
  int nbase = pbase & (NPT - 1);
  for (int e = tid; e < 8192; e += 256) {   // transpose (c,n) -> point-major
    int c = e >> 7, i = e & 127;
    Xl[i * 64 + c] = (__bf16)feat[b * 64 * NPT + c * NPT + nbase + i];
  }
  __syncthreads();

  int wv = tid >> 5, lane = tid & 31;
  v8f acc[4];
  zero_acc4(acc);
  gemm_64x64x16(Wl, Xl + wv * 16 * 64, lane, acc);
  int col = lane & 15, h = lane >> 4;
  int pt  = pbase + wv * 16 + col;
#pragma unroll
  for (int t = 0; t < 4; ++t)
#pragma unroll
    for (int r = 0; r < 8; ++r) {
      int   c = t * 16 + r + h * 8;
      float v = acc[t][r] * sS[c] + shS[c];
      outPM[pt * 64 + c] = fmaxf(v, 0.0f);
    }
}

// ---------------------------------------------------------------------------
// qkv: stacked [q(32);k(32);v(64)] x 64 GEMM. Q reads P1 tile, K/V read P2 tile.
// ---------------------------------------------------------------------------
__global__ void __launch_bounds__(256)
qkv_kernel(const float* __restrict__ P1, const float* __restrict__ P2,
           const float* __restrict__ qw, const float* __restrict__ qb,
           const float* __restrict__ kwp, const float* __restrict__ kbp,
           const float* __restrict__ vw, const float* __restrict__ vb,
           float* __restrict__ XQ, float* __restrict__ XK, float* __restrict__ XV) {
  __shared__ alignas(32) __bf16 Wl[128 * 64];
  __shared__ alignas(32) __bf16 Xa[128 * 64];
  __shared__ alignas(32) __bf16 Xb[128 * 64];
  __shared__ float bS[128];
  int tid = threadIdx.x;
  for (int e = tid; e < 2048; e += 256) {
    Wl[e]        = (__bf16)qw[e];
    Wl[2048 + e] = (__bf16)kwp[e];
  }
  for (int e = tid; e < 4096; e += 256) Wl[4096 + e] = (__bf16)vw[e];
  if (tid < 32) { bS[tid] = qb[tid]; bS[32 + tid] = kbp[tid]; }
  if (tid < 64) bS[64 + tid] = vb[tid];
  int pbase = blockIdx.x * 128;
  for (int e = tid; e < 8192; e += 256) {
    Xa[e] = (__bf16)P1[(size_t)pbase * 64 + e];
    Xb[e] = (__bf16)P2[(size_t)pbase * 64 + e];
  }
  __syncthreads();

  int wv = tid >> 5, lane = tid & 31;
  const __bf16* XaW = Xa + wv * 1024;
  const __bf16* XbW = Xb + wv * 1024;
  v16bf ba0 = load_bfrag(XaW, 0, lane), ba1 = load_bfrag(XaW, 1, lane);
  v16bf bb0 = load_bfrag(XbW, 0, lane), bb1 = load_bfrag(XbW, 1, lane);
  v8f acc[8];
#pragma unroll
  for (int t = 0; t < 8; ++t)
#pragma unroll
    for (int r = 0; r < 8; ++r) acc[t][r] = 0.0f;
#pragma unroll
  for (int t = 0; t < 8; ++t) {
    v16bf a0 = load_afrag(Wl, t, 0, lane);
    v16bf a1 = load_afrag(Wl, t, 1, lane);
    acc[t] = wmma_bf16(a0, (t < 2) ? ba0 : bb0, acc[t]);
    acc[t] = wmma_bf16(a1, (t < 2) ? ba1 : bb1, acc[t]);
  }
  int col = lane & 15, h = lane >> 4;
  int pt  = pbase + wv * 16 + col;
#pragma unroll
  for (int t = 0; t < 8; ++t)
#pragma unroll
    for (int r = 0; r < 8; ++r) {
      int   m = t * 16 + r + h * 8;
      float v = acc[t][r] + bS[m];
      if (t < 2)       XQ[pt * 32 + m] = v;
      else if (t < 4)  XK[pt * 32 + (m - 32)] = v;
      else             XV[pt * 64 + (m - 64)] = v;
    }
}

// ---------------------------------------------------------------------------
// cross: full transformer + cross epilogue. One wave per query point
// (16 neighbors = 16 WMMA columns). Two mlp1 WMMA layers, max over K via shfl.
// ---------------------------------------------------------------------------
__global__ void __launch_bounds__(128)
cross_kernel(const float* __restrict__ pcq, const float* __restrict__ pcr,
             const float* __restrict__ P1, const int* __restrict__ idx,
             const float* __restrict__ XQ, const float* __restrict__ XK,
             const float* __restrict__ XV,
             const float* __restrict__ p1w, const float* __restrict__ p1b,
             const float* __restrict__ pbng, const float* __restrict__ pbnb,
             const float* __restrict__ p2w, const float* __restrict__ p2b,
             const float* __restrict__ wbn1g, const float* __restrict__ wbn1b,
             const float* __restrict__ w1w, const float* __restrict__ w1b,
             const float* __restrict__ wbn2g, const float* __restrict__ wbn2b,
             const float* __restrict__ w2w, const float* __restrict__ w2b,
             const float* __restrict__ bt1g, const float* __restrict__ bt1b,
             const float* __restrict__ posw, const float* __restrict__ posb,
             const float* __restrict__ m1w1, const float* __restrict__ m1b1,
             const float* __restrict__ m1w2, const float* __restrict__ m1b2,
             float* __restrict__ outCM, float* __restrict__ outPM) {
  __shared__ alignas(32) __bf16 W1l[64 * 64];
  __shared__ alignas(32) __bf16 W2l[64 * 64];
  __shared__ float prS[4][16][64];
  __shared__ alignas(32) __bf16 Yl[4][16][64];
  __shared__ float wlS[4][16][8];
  __shared__ float p2wS[192], p2bS[64], poswS[192], posbS[64];
  __shared__ float bt1s[64], bt1sh[64], m1b1S[64], m1b2S[64];
  __shared__ float p1wS[9], pbS[3], pbSh[3];
  __shared__ float w1wS[128], wb1s[32], wb1sh[32];
  __shared__ float w2s[4], w2sh[4], w2wS[32], w2bS[8];

  int tid = threadIdx.x;
  const float rs = rsqrtf(1.0f + EPSV);
  for (int e = tid; e < 4096; e += 128) {
    W1l[e] = (__bf16)m1w1[e];
    W2l[e] = (__bf16)m1w2[e];
  }
  for (int e = tid; e < 192; e += 128) { p2wS[e] = p2w[e]; poswS[e] = posw[e]; }
  if (tid < 64) {
    p2bS[tid]  = p2b[tid];
    posbS[tid] = posb[tid];
    bt1s[tid]  = bt1g[tid] * rs;
    bt1sh[tid] = bt1b[tid];
    m1b1S[tid] = m1b1[tid];
    m1b2S[tid] = m1b2[tid];
  }
  if (tid < 9) p1wS[tid] = p1w[tid];
  if (tid < 3) {
    float s = pbng[tid] * rs;
    pbS[tid]  = s;
    pbSh[tid] = p1b[tid] * s + pbnb[tid];
  }
  if (tid < 128) w1wS[tid] = w1w[tid];
  if (tid < 32) {
    wb1s[tid]  = wbn1g[tid] * rs;
    wb1sh[tid] = wbn1b[tid];
    w2wS[tid]  = w2w[tid];
  }
  if (tid < 4) {
    float s2 = wbn2g[tid] * rs;
    w2s[tid]  = s2;
    w2sh[tid] = w1b[tid] * s2 + wbn2b[tid];
  }
  if (tid < 8) w2bS[tid] = w2b[tid];
  __syncthreads();

  int w = tid >> 5, lane = tid & 31;
  int k = lane & 15, h = lane >> 4;
  int p = blockIdx.x * 4 + w;               // global query point
  int b = p >> 12, n = p & (NPT - 1);

  int jloc = idx[p * 16 + k];
  int jg   = b * NPT + jloc;
  float qx = pcq[b * 3 * NPT + n];
  float qy = pcq[b * 3 * NPT + NPT + n];
  float qz = pcq[b * 3 * NPT + 2 * NPT + n];
  float dx = pcr[b * 3 * NPT + jloc] - qx;
  float dy = pcr[b * 3 * NPT + NPT + jloc] - qy;
  float dz = pcr[b * 3 * NPT + 2 * NPT + jloc] - qz;

  // positional MLP 3 -> 3 (bn+relu) -> 64
  float u0, u1, u2;
  {
    float L0 = p1wS[0] * dx + p1wS[1] * dy + p1wS[2] * dz;
    float L1 = p1wS[3] * dx + p1wS[4] * dy + p1wS[5] * dz;
    float L2 = p1wS[6] * dx + p1wS[7] * dy + p1wS[8] * dz;
    u0 = fmaxf(L0 * pbS[0] + pbSh[0], 0.0f);
    u1 = fmaxf(L1 * pbS[1] + pbSh[1], 0.0f);
    u2 = fmaxf(L2 * pbS[2] + pbSh[2], 0.0f);
  }
  int c0 = h * 32;
  for (int cc = 0; cc < 32; ++cc) {
    int c = c0 + cc;
    prS[w][k][c] = p2wS[c * 3] * u0 + p2wS[c * 3 + 1] * u1 + p2wS[c * 3 + 2] * u2 +
                   p2bS[c];
  }

  // attention-weight pipeline: (gk - xq + pr0+pr1) -> bn relu -> 32->4 -> bn relu -> 4->8
  float a0 = 0.f, a1 = 0.f, a2 = 0.f, a3 = 0.f;
  for (int c = 0; c < 32; ++c) {
    float wv = XK[jg * 32 + c] - XQ[p * 32 + c] + prS[w][k][c] + prS[w][k][32 + c];
    wv = fmaxf(wv * wb1s[c] + wb1sh[c], 0.0f);
    a0 += w1wS[c] * wv;
    a1 += w1wS[32 + c] * wv;
    a2 += w1wS[64 + c] * wv;
    a3 += w1wS[96 + c] * wv;
  }
  float h0 = fmaxf(a0 * w2s[0] + w2sh[0], 0.0f);
  float h1 = fmaxf(a1 * w2s[1] + w2sh[1], 0.0f);
  float h2 = fmaxf(a2 * w2s[2] + w2sh[2], 0.0f);
  float h3 = fmaxf(a3 * w2s[3] + w2sh[3], 0.0f);
  if (h == 0) {
#pragma unroll
    for (int m = 0; m < 8; ++m)
      wlS[w][k][m] = w2wS[m * 4] * h0 + w2wS[m * 4 + 1] * h1 + w2wS[m * 4 + 2] * h2 +
                     w2wS[m * 4 + 3] * h3 + w2bS[m];
  }
  // softmax over the 16 neighbors, lanes 0..7 each own one of the 8 channels
  if (lane < 8) {
    int   m  = lane;
    float mx = -3.0e38f;
#pragma unroll
    for (int kk = 0; kk < 16; ++kk) mx = fmaxf(mx, wlS[w][kk][m]);
    float ex[16];
    float s = 0.0f;
#pragma unroll
    for (int kk = 0; kk < 16; ++kk) {
      float e = __expf(wlS[w][kk][m] - mx);
      ex[kk]  = e;
      s += e;
    }
    float inv = 1.0f / s;
#pragma unroll
    for (int kk = 0; kk < 16; ++kk) wlS[w][kk][m] = ex[kk] * inv;
  }

  // build Y(16 neighbors x 64 chan): relu(relu(bn_t1((gv+pr)*w)) + gp1 + dirf)
  for (int cc = 0; cc < 32; ++cc) {
    int   c = c0 + cc;
    float t = (XV[jg * 64 + c] + prS[w][k][c]) * wlS[w][k][c & 7];
    t = fmaxf(t * bt1s[c] + bt1sh[c], 0.0f);
    float y = t + P1[p * 64 + c] +
              (poswS[c * 3] * dx + poswS[c * 3 + 1] * dy + poswS[c * 3 + 2] * dz +
               posbS[c]);
    Yl[w][k][c] = (__bf16)fmaxf(y, 0.0f);
  }

  // mlp1 layer 1 (WMMA) + leaky, write back to Yl
  v8f acc[4];
  zero_acc4(acc);
  gemm_64x64x16(W1l, &Yl[w][0][0], lane, acc);
#pragma unroll
  for (int t = 0; t < 4; ++t)
#pragma unroll
    for (int r = 0; r < 8; ++r) {
      int   c = t * 16 + r + h * 8;
      float v = acc[t][r] + m1b1S[c];
      v = (v > 0.0f) ? v : LEAKV * v;
      Yl[w][lane & 15][c] = (__bf16)v;
    }

  // mlp1 layer 2 (WMMA) + leaky, then max over the 16 neighbor columns
  v8f acc2[4];
  zero_acc4(acc2);
  gemm_64x64x16(W2l, &Yl[w][0][0], lane, acc2);
#pragma unroll
  for (int t = 0; t < 4; ++t)
#pragma unroll
    for (int r = 0; r < 8; ++r) {
      int   c = t * 16 + r + h * 8;
      float v = acc2[t][r] + m1b2S[c];
      v = (v > 0.0f) ? v : LEAKV * v;
      v = fmaxf(v, __shfl_xor(v, 1));
      v = fmaxf(v, __shfl_xor(v, 2));
      v = fmaxf(v, __shfl_xor(v, 4));
      v = fmaxf(v, __shfl_xor(v, 8));
      if ((lane & 15) == 0) {
        outCM[b * 64 * NPT + c * NPT + n] = v;
        outPM[p * 64 + c]                 = v;
      }
    }
}

// ---------------------------------------------------------------------------
// old_cross: 4 points x 4 neighbors per wave = 16 WMMA columns; one mlp layer.
// ---------------------------------------------------------------------------
__global__ void __launch_bounds__(128)
oldcross_kernel(const float* __restrict__ pcq, const float* __restrict__ pcr,
                const float* __restrict__ F1PM, const float* __restrict__ F2PM,
                const int* __restrict__ idx, const float* __restrict__ posw,
                const float* __restrict__ posb, const float* __restrict__ mw,
                const float* __restrict__ mb, float* __restrict__ outCM) {
  __shared__ alignas(32) __bf16 Wl[64 * 64];
  __shared__ alignas(32) __bf16 Yl[4][16][64];
  __shared__ float poswS[192], posbS[64], mbS[64];
  int tid = threadIdx.x;
  for (int e = tid; e < 4096; e += 128) Wl[e] = (__bf16)mw[e];
  for (int e = tid; e < 192; e += 128) poswS[e] = posw[e];
  if (tid < 64) { posbS[tid] = posb[tid]; mbS[tid] = mb[tid]; }
  __syncthreads();

  int wv = tid >> 5, lane = tid & 31;
  int col = lane & 15, h = lane >> 4;
  int pt4 = col >> 2, k = col & 3;
  int p = blockIdx.x * 16 + wv * 4 + pt4;
  int b = p >> 12, n = p & (NPT - 1);
  int jloc = idx[p * 16 + k];       // first 4 of sorted top-16 == top-4
  int jg   = b * NPT + jloc;
  float qx = pcq[b * 3 * NPT + n];
  float qy = pcq[b * 3 * NPT + NPT + n];
  float qz = pcq[b * 3 * NPT + 2 * NPT + n];
  float dx = pcr[b * 3 * NPT + jloc] - qx;
  float dy = pcr[b * 3 * NPT + NPT + jloc] - qy;
  float dz = pcr[b * 3 * NPT + 2 * NPT + jloc] - qz;

  int c0 = h * 32;
  for (int cc = 0; cc < 32; ++cc) {
    int   c = c0 + cc;
    float y = F2PM[jg * 64 + c] + F1PM[p * 64 + c] + poswS[c * 3] * dx +
              poswS[c * 3 + 1] * dy + poswS[c * 3 + 2] * dz + posbS[c];
    Yl[wv][col][c] = (__bf16)fmaxf(y, 0.0f);
  }

  v8f acc[4];
  zero_acc4(acc);
  gemm_64x64x16(Wl, &Yl[wv][0][0], lane, acc);
#pragma unroll
  for (int t = 0; t < 4; ++t)
#pragma unroll
    for (int r = 0; r < 8; ++r) {
      int   c = t * 16 + r + h * 8;
      float v = acc[t][r] + mbS[c];
      v = (v > 0.0f) ? v : LEAKV * v;
      v = fmaxf(v, __shfl_xor(v, 1));
      v = fmaxf(v, __shfl_xor(v, 2));
      if ((lane & 3) == 0) outCM[b * 64 * NPT + c * NPT + n] = v;
    }
}

// ---------------------------------------------------------------------------
// Host launcher
// ---------------------------------------------------------------------------
enum {
  IN_PC1 = 0, IN_PC2, IN_FEAT1, IN_FEAT2,
  P_T11_W, P_T11_B, P_T22_W, P_T22_B,
  P_BN1_G, P_BN1_B, P_BN2_G, P_BN2_B,
  P_BNT1_G, P_BNT1_B,
  P_POS1_W, P_POS1_B, P_POS2_W, P_POS2_B, P_POSOLD_W, P_POSOLD_B,
  P_Q_W, P_Q_B, P_K_W, P_K_B, P_V_W, P_V_B,
  P_P1_W, P_P1_B, P_PBN_G, P_PBN_B, P_P2_W, P_P2_B,
  P_WBN1_G, P_WBN1_B, P_W1_W, P_W1_B, P_WBN2_G, P_WBN2_B, P_W2_W, P_W2_B,
  P_M1_W1, P_M1_B1, P_M1_W2, P_M1_B2,
  P_MOLD_W, P_MOLD_B
};

extern "C" void kernel_launch(void* const* d_in, const int* in_sizes, int n_in,
                              void* d_out, int out_size, void* d_ws, size_t ws_size,
                              hipStream_t stream) {
  (void)in_sizes; (void)n_in; (void)out_size; (void)ws_size;
  auto F = [&](int i) { return (const float*)d_in[i]; };
  float* out = (float*)d_out;
  float* ws  = (float*)d_ws;

  float* PA   = ws;                      // relu(bn1(t11(feat1)))   [NPALL,64]
  float* PB   = PA + NPALL * 64;         // relu(bn2(t22(feat2)))
  float* PC   = PB + NPALL * 64;         // relu(bn1(t11(feat2)))
  float* PD   = PC + NPALL * 64;         // relu(bn2(t22(feat1)))
  float* XQ1  = PD + NPALL * 64;
  float* XK1  = XQ1 + NPALL * 32;
  float* XV1  = XK1 + NPALL * 32;
  float* XQ2  = XV1 + NPALL * 64;
  float* XK2  = XQ2 + NPALL * 32;
  float* XV2  = XK2 + NPALL * 32;
  int*   IDX1 = (int*)(XV2 + NPALL * 64);
  int*   IDX2 = IDX1 + NPALL * 16;
  float* F1PM = (float*)(IDX2 + NPALL * 16);
  float* F2PM = F1PM + NPALL * 64;

  knn16_kernel<<<NPALL / 256, 256, 0, stream>>>(F(IN_PC1), F(IN_PC2), IDX1);
  knn16_kernel<<<NPALL / 256, 256, 0, stream>>>(F(IN_PC2), F(IN_PC1), IDX2);

  tlinear_kernel<<<NPALL / 128, 256, 0, stream>>>(F(IN_FEAT1), F(P_T11_W), F(P_T11_B),
                                                  F(P_BN1_G), F(P_BN1_B), PA);
  tlinear_kernel<<<NPALL / 128, 256, 0, stream>>>(F(IN_FEAT2), F(P_T22_W), F(P_T22_B),
                                                  F(P_BN2_G), F(P_BN2_B), PB);
  tlinear_kernel<<<NPALL / 128, 256, 0, stream>>>(F(IN_FEAT2), F(P_T11_W), F(P_T11_B),
                                                  F(P_BN1_G), F(P_BN1_B), PC);
  tlinear_kernel<<<NPALL / 128, 256, 0, stream>>>(F(IN_FEAT1), F(P_T22_W), F(P_T22_B),
                                                  F(P_BN2_G), F(P_BN2_B), PD);

  qkv_kernel<<<NPALL / 128, 256, 0, stream>>>(PA, PB, F(P_Q_W), F(P_Q_B), F(P_K_W),
                                              F(P_K_B), F(P_V_W), F(P_V_B), XQ1, XK1,
                                              XV1);
  qkv_kernel<<<NPALL / 128, 256, 0, stream>>>(PC, PD, F(P_Q_W), F(P_Q_B), F(P_K_W),
                                              F(P_K_B), F(P_V_W), F(P_V_B), XQ2, XK2,
                                              XV2);

  cross_kernel<<<NPALL / 4, 128, 0, stream>>>(
      F(IN_PC1), F(IN_PC2), PA, IDX1, XQ1, XK1, XV1, F(P_P1_W), F(P_P1_B),
      F(P_PBN_G), F(P_PBN_B), F(P_P2_W), F(P_P2_B), F(P_WBN1_G), F(P_WBN1_B),
      F(P_W1_W), F(P_W1_B), F(P_WBN2_G), F(P_WBN2_B), F(P_W2_W), F(P_W2_B),
      F(P_BNT1_G), F(P_BNT1_B), F(P_POS1_W), F(P_POS1_B), F(P_M1_W1), F(P_M1_B1),
      F(P_M1_W2), F(P_M1_B2), out, F1PM);
  cross_kernel<<<NPALL / 4, 128, 0, stream>>>(
      F(IN_PC2), F(IN_PC1), PC, IDX2, XQ2, XK2, XV2, F(P_P1_W), F(P_P1_B),
      F(P_PBN_G), F(P_PBN_B), F(P_P2_W), F(P_P2_B), F(P_WBN1_G), F(P_WBN1_B),
      F(P_W1_W), F(P_W1_B), F(P_WBN2_G), F(P_WBN2_B), F(P_W2_W), F(P_W2_B),
      F(P_BNT1_G), F(P_BNT1_B), F(P_POS2_W), F(P_POS2_B), F(P_M1_W1), F(P_M1_B1),
      F(P_M1_W2), F(P_M1_B2), out + NPALL * 64, F2PM);

  oldcross_kernel<<<NPALL / 16, 128, 0, stream>>>(
      F(IN_PC1), F(IN_PC2), F1PM, F2PM, IDX1, F(P_POSOLD_W), F(P_POSOLD_B),
      F(P_MOLD_W), F(P_MOLD_B), out + 2 * NPALL * 64);
}
